// DN4Metric_19396072309313
// MI455X (gfx1250) — matrix operand order repaired and checked by reading, not compiled, not run on hardware
//
#include <hip/hip_runtime.h>
#include <hip/hip_bf16.h>
#include <math.h>

// Problem constants (fixed by reference setup_inputs)
#define BS 2
#define NW 5
#define NS 5
#define NQ 5
#define FD 64          // channel dim (K)
#define PIX 441        // 21*21 pixels
#define PPAD 448       // padded to multiple of 16
#define NIMG (BS*NW*(NS+NQ))   // 100
#define NTILES (PPAD/16)       // 28 tiles per image
#define STILES (NS*NTILES)     // 140 support tiles per (b,w)
#define NPRED (BS*NW*NQ*NW)    // 250
#define NEL (BS*NW*NQ)         // 50
#define NEGINF (-3.0e38f)

typedef _Float16 v8h  __attribute__((ext_vector_type(8)));
typedef _Float16 v16h __attribute__((ext_vector_type(16)));
typedef float    v8f  __attribute__((ext_vector_type(8)));

// Branchless top-3 insert: 5 v_max_num/v_min_num, no branches.
__device__ __forceinline__ void top3_insert(float v, float& t0, float& t1, float& t2) {
    float n0 = fmaxf(t0, v);
    float l0 = fminf(t0, v);
    float n1 = fmaxf(t1, l0);
    float l1 = fminf(t1, l0);
    float n2 = fmaxf(t2, l1);
    t0 = n0; t1 = n1; t2 = n2;
}

// ---------------------------------------------------------------------------
// Kernel 0: zero pred region of d_out, write query labels (as float)
// ---------------------------------------------------------------------------
__global__ void dn4_init_out(const int* __restrict__ elabel, float* __restrict__ out) {
    int i = threadIdx.x;
    if (i < NPRED) out[i] = 0.0f;
    if (i < NEL) {
        int iq = i % NQ;
        int w  = (i / NQ) % NW;
        int b  = i / (NQ * NW);
        int src = (b * NW + w) * (NS + NQ) + NS + iq;
        out[NPRED + i] = (float)elabel[src];
    }
}

// ---------------------------------------------------------------------------
// Kernel 1: L2-normalize each 64-dim pixel descriptor, convert to f16,
// pack as [img][pixel_pad=448][64] (channel-contiguous), zero-pad pixels.
// ---------------------------------------------------------------------------
__global__ void dn4_normalize(const float* __restrict__ fm, _Float16* __restrict__ desc) {
    int idx = blockIdx.x * blockDim.x + threadIdx.x;
    if (idx >= NIMG * PPAD) return;
    int img = idx / PPAD;
    int pix = idx - img * PPAD;
    _Float16* out = desc + (size_t)idx * FD;
    if (pix < PIX) {
        const float* src = fm + (size_t)img * FD * PIX + pix;
        float v[FD];
        float s = 0.0f;
#pragma unroll
        for (int c = 0; c < FD; ++c) { v[c] = src[c * PIX]; s += v[c] * v[c]; }
        float scale = 1.0f / (sqrtf(s) + 1e-12f);
#pragma unroll
        for (int c = 0; c < FD; ++c) out[c] = (_Float16)(v[c] * scale);
    } else {
#pragma unroll
        for (int c = 0; c < FD; ++c) out[c] = (_Float16)0.0f;
    }
}

// ---------------------------------------------------------------------------
// Kernel 2: one wave32 per (b, q, w, x-tile-pair).
//   A = support tile (M = 16 y-pixels, streamed over 140 contiguous tiles,
//       double-buffered), B = que pixels (N = 2x16 x-pixels, register-resident).
//   C layout puts 8 y-candidates per lane for the lane's own x column, so
//   top-3 accumulates branch-free in registers. Halves merged via shfl_xor.
// ---------------------------------------------------------------------------
__global__ __launch_bounds__(32)
void dn4_gemm_topk(const _Float16* __restrict__ desc, float* __restrict__ pred) {
    const int xp = blockIdx.x;                 // x tile pair: 0..13
    const int w  = blockIdx.y;                 // class:      0..4
    const int bq = blockIdx.z;                 // 0..49
    const int b  = bq / (NW * NQ);
    const int q  = bq - b * (NW * NQ);         // 0..24
    const int wq = q / NQ;
    const int iq = q - wq * NQ;
    const int img_q = (b * NW + wq) * (NS + NQ) + NS + iq;

    const int lane = threadIdx.x;
    const int ln = lane & 15;
    const int hi = lane >> 4;

    // --- B fragments (que), 32x16 f16 layout: lane column n=ln,
    //     16 contiguous halves at k = kk + hi*16.  Loaded once. ---
    v16h b0[2], b1[2];
#pragma unroll
    for (int j = 0; j < 2; ++j) {
        const _Float16* bp =
            desc + ((size_t)img_q * PPAD + (xp * 2 + j) * 16 + ln) * FD + hi * 16;
        b0[j] = *(const v16h*)(bp);
        b1[j] = *(const v16h*)(bp + 32);
    }

    // 5 support images of (b,w) are contiguous: 140 contiguous 16-row tiles.
    const _Float16* Sbase =
        desc + (size_t)((b * NW + w) * (NS + NQ)) * PPAD * FD;

    // --- A fragment (support), 16x32 f16 layout: lane row m=ln,
    //     16 contiguous halves at k = kk + hi*16 ---
    auto loadA = [&](int t, v16h& A0, v16h& A1) {
        const _Float16* ap = Sbase + ((size_t)t * 16 + ln) * FD + hi * 16;
        v8h p0 = *(const v8h*)(ap);
        v8h p1 = *(const v8h*)(ap + 8);
        v8h p2 = *(const v8h*)(ap + 32);
        v8h p3 = *(const v8h*)(ap + 40);
#pragma unroll
        for (int i = 0; i < 8; ++i) {
            A0[i] = p0[i]; A0[8 + i] = p1[i];
            A1[i] = p2[i]; A1[8 + i] = p3[i];
        }
    };

    float t0[2] = {NEGINF, NEGINF};
    float t1[2] = {NEGINF, NEGINF};
    float t2[2] = {NEGINF, NEGINF};

    v16h a0, a1;
    loadA(0, a0, a1);
    int ntp = 0;                               // pixel offset of tile within image

#pragma unroll 2
    for (int t = 0; t < STILES; ++t) {
        // Prefetch next support tile (double buffer).
        v16h na0, na1;
        loadA(t + 1 < STILES ? t + 1 : t, na0, na1);

        v8f c0 = {0.f, 0.f, 0.f, 0.f, 0.f, 0.f, 0.f, 0.f};
        v8f c1 = {0.f, 0.f, 0.f, 0.f, 0.f, 0.f, 0.f, 0.f};
        c0 = __builtin_amdgcn_wmma_f32_16x16x32_f16(false, a0, false, b0[0],
                                                    (short)0, c0, false, false);
        c0 = __builtin_amdgcn_wmma_f32_16x16x32_f16(false, a1, false, b1[0],
                                                    (short)0, c0, false, false);
        c1 = __builtin_amdgcn_wmma_f32_16x16x32_f16(false, a0, false, b0[1],
                                                    (short)0, c1, false, false);
        c1 = __builtin_amdgcn_wmma_f32_16x16x32_f16(false, a1, false, b1[1],
                                                    (short)0, c1, false, false);

        if (ntp == (NTILES - 1) * 16) {
            // Tail tile of an image: rows m >= PIX-ntp (=9) are zero padding.
#pragma unroll
            for (int v = 0; v < 8; ++v) {
                bool ok = (8 * hi + v) < (PIX - (NTILES - 1) * 16);
                top3_insert(ok ? c0[v] : NEGINF, t0[0], t1[0], t2[0]);
                top3_insert(ok ? c1[v] : NEGINF, t0[1], t1[1], t2[1]);
            }
        } else {
#pragma unroll
            for (int v = 0; v < 8; ++v) {
                top3_insert(c0[v], t0[0], t1[0], t2[0]);
                top3_insert(c1[v], t0[1], t1[1], t2[1]);
            }
        }

        a0 = na0; a1 = na1;
        ntp += 16;
        if (ntp == PPAD) ntp = 0;
    }

    // Merge the two hi-halves (disjoint y-rows, same x column), then reduce.
#pragma unroll
    for (int j = 0; j < 2; ++j) {
        float o0 = __shfl_xor(t0[j], 16, 32);
        float o1 = __shfl_xor(t1[j], 16, 32);
        float o2 = __shfl_xor(t2[j], 16, 32);
        top3_insert(o0, t0[j], t1[j], t2[j]);
        top3_insert(o1, t0[j], t1[j], t2[j]);
        top3_insert(o2, t0[j], t1[j], t2[j]);
        if (lane < 16) {
            int x = (xp * 2 + j) * 16 + ln;
            if (x < PIX)
                atomicAdd(&pred[(b * (NW * NQ) + q) * NW + w],
                          t0[j] + t1[j] + t2[j]);
        }
    }
}

// ---------------------------------------------------------------------------
extern "C" void kernel_launch(void* const* d_in, const int* in_sizes, int n_in,
                              void* d_out, int out_size, void* d_ws, size_t ws_size,
                              hipStream_t stream) {
    const float* fm     = (const float*)d_in[0];
    const int*   elabel = (const int*)d_in[1];
    // d_in[2] (glabel) and scalar shape args unused (shapes fixed by setup).

    _Float16* desc = (_Float16*)d_ws;            // NIMG*PPAD*FD f16 ~ 5.5 MB
    float*    out  = (float*)d_out;              // 250 pred + 50 labels

    dn4_init_out<<<1, 256, 0, stream>>>(elabel, out);

    int total = NIMG * PPAD;
    dn4_normalize<<<(total + 255) / 256, 256, 0, stream>>>(fm, desc);

    dim3 grid(NTILES / 2, NW, BS * NW * NQ);     // (14, 5, 50)
    dn4_gemm_topk<<<grid, 32, 0, stream>>>(desc, out);
}
